// MultiHeadedAttendtion_70179765616863
// MI455X (gfx1250) — matrix operand order repaired and checked by reading, not compiled, not run on hardware
//
#include <hip/hip_runtime.h>

// MI455X / gfx1250, wave32. Multi-head attention forward.
// Pipeline:
//   0) convert W{q,k,v,o} f32->f16 (once)
//   1) 3x QKV projection GEMM (A: f32 convert-stage, W: async-to-LDS, double-buffered)
//   2) flash attention w/ softmax-1 (K staged via TDM tensor_load_to_lds with LDS row
//      padding, double-buffered; V transpose-staged through VGPRs) -> f16 X
//   3) output projection GEMM (A=X f16 async-to-LDS, W async-to-LDS) -> f32 d_out
// All matmuls on v_wmma_f32_16x16x32_f16.

typedef __attribute__((ext_vector_type(16))) _Float16 v16h;
typedef __attribute__((ext_vector_type(8)))  _Float16 v8h;
typedef __attribute__((ext_vector_type(8)))  float    v8f;
typedef __attribute__((ext_vector_type(4)))  unsigned u32x4;
typedef __attribute__((ext_vector_type(4)))  int      i32x4;
typedef __attribute__((ext_vector_type(8)))  int      i32x8;

#define WMMA_F32_F16(A_, B_, C_) \
  __builtin_amdgcn_wmma_f32_16x16x32_f16(false, (A_), false, (B_), (short)0, (C_), false, false)

static constexpr int Bc  = 2;
static constexpr int Sc  = 2048;
static constexpr int Dc  = 1024;
static constexpr int Hc  = 16;
static constexpr int DKc = 64;

// ---------------------------------------------------------------------------
// Async global->LDS 16-byte copy (GVS mode) + ASYNCcnt fences.
// ---------------------------------------------------------------------------
__device__ __forceinline__ void async_copy_b128(void* lds_ptr, const void* gbase,
                                                unsigned byte_off) {
    unsigned lds_off = (unsigned)(unsigned long long)lds_ptr;   // low 32 bits = LDS offset
    asm volatile("global_load_async_to_lds_b128 %0, %1, %2"
                 :: "v"(lds_off), "v"(byte_off),
                    "s"((unsigned long long)gbase)
                 : "memory");
}
__device__ __forceinline__ void async_wait0() { asm volatile("s_wait_asynccnt 0x0" ::: "memory"); }
__device__ __forceinline__ void async_wait2() { asm volatile("s_wait_asynccnt 0x2" ::: "memory"); }
__device__ __forceinline__ void async_wait3() { asm volatile("s_wait_asynccnt 0x3" ::: "memory"); }

// ---------------------------------------------------------------------------
// Tensor Data Mover: 2D f16 tile (tile_d1 rows x tile_d0 elems) global->LDS,
// with optional LDS row padding. Descriptor per CDNA5 ISA D# groups 0/1.
// pad_interval code: threshold = (2 << code) DWORDs; pad_amount code: (code+1) DWORDs.
// ---------------------------------------------------------------------------
__device__ __forceinline__ void tdm_load_2d_f16(unsigned lds_byte_addr,
                                                const void* tile_gaddr,
                                                unsigned tensor_d0, unsigned tensor_d1,
                                                unsigned tile_d0,   unsigned tile_d1,
                                                unsigned d0_stride,
                                                unsigned pad_interval_code,
                                                unsigned pad_amount_code,
                                                bool pad_en)
{
    unsigned long long ga = (unsigned long long)tile_gaddr;
    u32x4 g0;
    g0[0] = 1u;                                           // count=1, user descriptor
    g0[1] = lds_byte_addr;                                // lds_addr
    g0[2] = (unsigned)(ga & 0xFFFFFFFFu);                 // global_addr[31:0]
    g0[3] = (unsigned)((ga >> 32) & 0x01FFFFFFu) | (2u << 30);   // ga[56:32] | type=2
    i32x8 g1;
    g1[0] = (int)((1u << 16)                              // data_size = 2 bytes
                | (pad_en ? (1u << 20) : 0u)              // pad_enable
                | (pad_interval_code << 22)
                | (pad_amount_code << 25));
    g1[1] = (int)(tensor_d0 << 16);                       // tensor_dim0[15:0] @ bit48
    g1[2] = (int)((tensor_d0 >> 16) | (tensor_d1 << 16)); // dim0 hi | dim1 lo
    g1[3] = (int)((tensor_d1 >> 16) | (tile_d0 << 16));   // dim1 hi | tile_dim0
    g1[4] = (int)(tile_d1);                               // tile_dim1 (tile_dim2=0)
    g1[5] = (int)(d0_stride);                             // tensor_dim0_stride[31:0]
    g1[6] = 0;                                            // stride hi | dim1_stride lo
    g1[7] = 0;
    i32x4 z4 = {0, 0, 0, 0};
#if __clang_major__ >= 23
    i32x8 z8 = {0, 0, 0, 0, 0, 0, 0, 0};
    __builtin_amdgcn_tensor_load_to_lds(g0, g1, z4, z4, z8, 0);
#else
    __builtin_amdgcn_tensor_load_to_lds(g0, g1, z4, z4, 0);
#endif
}

// ---------------------------------------------------------------------------
// f32 -> f16 elementwise convert (weights), 8 elems/thread.
// ---------------------------------------------------------------------------
__global__ __launch_bounds__(256)
void cvt_f32_to_f16_kernel(const float* __restrict__ src, _Float16* __restrict__ dst, int n)
{
    int i = (blockIdx.x * 256 + threadIdx.x) * 8;
    if (i + 7 < n) {
        float4 f0 = ((const float4*)(src + i))[0];
        float4 f1 = ((const float4*)(src + i))[1];
        v8h h;
        h[0] = (_Float16)f0.x; h[1] = (_Float16)f0.y;
        h[2] = (_Float16)f0.z; h[3] = (_Float16)f0.w;
        h[4] = (_Float16)f1.x; h[5] = (_Float16)f1.y;
        h[6] = (_Float16)f1.z; h[7] = (_Float16)f1.w;
        *(v8h*)(dst + i) = h;
    }
}

// ---------------------------------------------------------------------------
// Projection GEMM: C[m,n] = sum_k A[m,k] * W[n,k] + bias[n]
//   A: f32 (convert-stage) or f16 (async-stage), M x K row-major.
//   W: f16, N x K row-major, async-staged. Double-buffered K chunks of 64.
//   HEAD_STORE: f16 (B,H,S,DK) layout; else f32 row-major M x N.
// Block: 256 threads = 8 waves, tile 32(M) x 64(N).
// ---------------------------------------------------------------------------
template<bool A_IS_F16, bool HEAD_STORE>
__global__ __launch_bounds__(256)
void proj_gemm_kernel(const float* __restrict__ Af, const _Float16* __restrict__ Ah,
                      const _Float16* __restrict__ Wh, const float* __restrict__ bias,
                      _Float16* __restrict__ out_h, float* __restrict__ out_f,
                      int M, int N, int K)
{
    constexpr int AL = 72;   // LDS row strides (f16 elems), 16B-aligned rows
    constexpr int BL = 72;
    __shared__ alignas(16) _Float16 As[2][32 * AL];
    __shared__ alignas(16) _Float16 Ws[2][64 * BL];

    const int tid  = threadIdx.x;
    const int lane = tid & 31;
    const int wave = tid >> 5;
    const int wm   = wave >> 2;          // 0..1  (M sub-tile)
    const int wn   = wave & 3;           // 0..3  (N sub-tile)
    const int m0   = blockIdx.y * 32;
    const int n0   = blockIdx.x * 64;

    const int mloc = lane & 15;
    const int half = lane >> 4;

    // staging assignments
    const int ar = tid >> 3;             // A tile row (0..31), 8 elems each
    const int ac = (tid & 7) * 8;
    const int wr = tid >> 2;             // W tile row (0..63), 16 f16 each
    const int wc = (tid & 3) * 16;

    v8f acc = {};
    float4 af0, af1;

    // ---- preload chunk 0 ----
    {
        unsigned off = (unsigned)(((n0 + wr) * K + wc) * 2);
        async_copy_b128(&Ws[0][wr * BL + wc + 0], Wh, off);
        async_copy_b128(&Ws[0][wr * BL + wc + 8], Wh, off + 16);
    }
    if (A_IS_F16) {
        unsigned off = (unsigned)(((m0 + ar) * K + ac) * 2);
        async_copy_b128(&As[0][ar * AL + ac], Ah, off);
    } else {
        const float* src = Af + (size_t)(m0 + ar) * K + ac;
        af0 = ((const float4*)src)[0];
        af1 = ((const float4*)src)[1];
    }

    const int NCH = K / 64;
    for (int c = 0; c < NCH; ++c) {
        const int cur = c & 1;
        const int nxt = cur ^ 1;
        const int k0  = c * 64;

        // f32 path: commit staged A regs (chunk c) to LDS
        if (!A_IS_F16) {
            v8h h;
            h[0] = (_Float16)af0.x; h[1] = (_Float16)af0.y;
            h[2] = (_Float16)af0.z; h[3] = (_Float16)af0.w;
            h[4] = (_Float16)af1.x; h[5] = (_Float16)af1.y;
            h[6] = (_Float16)af1.z; h[7] = (_Float16)af1.w;
            *(v8h*)&As[cur][ar * AL + ac] = h;
        }
        // kick off chunk c+1 into the other buffer
        if (c + 1 < NCH) {
            unsigned woff = (unsigned)(((n0 + wr) * K + k0 + 64 + wc) * 2);
            async_copy_b128(&Ws[nxt][wr * BL + wc + 0], Wh, woff);
            async_copy_b128(&Ws[nxt][wr * BL + wc + 8], Wh, woff + 16);
            if (A_IS_F16) {
                unsigned aoff = (unsigned)(((m0 + ar) * K + k0 + 64 + ac) * 2);
                async_copy_b128(&As[nxt][ar * AL + ac], Ah, aoff);
            } else {
                const float* src = Af + (size_t)(m0 + ar) * K + k0 + 64 + ac;
                af0 = ((const float4*)src)[0];
                af1 = ((const float4*)src)[1];
            }
        }
        // wait for chunk c's asyncs only; chunk c+1's stay in flight
        if (c + 1 < NCH) { if (A_IS_F16) async_wait3(); else async_wait2(); }
        else             { async_wait0(); }
        __syncthreads();

        // ---- compute: two 16x16x32 WMMAs per K chunk ----
        #pragma unroll
        for (int kk = 0; kk < 64; kk += 32) {
            v16h a, b;
            const _Float16* arow = &As[cur][(wm * 16 + mloc) * AL + kk + half * 8];
            *((v8h*)&a + 0) = *(const v8h*)(arow + 0);   // K = kk+half*8 .. +7
            *((v8h*)&a + 1) = *(const v8h*)(arow + 16);  // K = kk+16+half*8 .. +7
            const _Float16* brow = &Ws[cur][(wn * 16 + mloc) * BL + kk + half * 8];
            *((v8h*)&b + 0) = *(const v8h*)(brow + 0);
            *((v8h*)&b + 1) = *(const v8h*)(brow + 16);
            acc = WMMA_F32_F16(a, b, acc);
        }
        __syncthreads();   // all reads of buffer `cur` done before it is refilled
    }

    // ---- bias + store (C layout: VGPR r -> M = half*8 + r, N = lane%16) ----
    const int ng  = n0 + wn * 16 + mloc;
    const float bv = bias[ng];
    const int mb  = m0 + wm * 16 + half * 8;
    #pragma unroll
    for (int r = 0; r < 8; ++r) {
        float cval = acc[r] + bv;
        int m = mb + r;
        if (HEAD_STORE) {
            int bb = m >> 11, ss = m & 2047;        // m = b*S + s (S=2048)
            int hh = ng >> 6, dd = ng & 63;         // n = h*DK + dk (DK=64)
            out_h[(((size_t)(bb * Hc + hh)) * Sc + ss) * DKc + dd] = (_Float16)cval;
        } else {
            out_f[(size_t)m * N + ng] = cval;
        }
    }
}

// ---------------------------------------------------------------------------
// Flash attention with softmax-1 (denominator = 1 + sum(exp(s - m))).
// Block: 128 threads = 4 waves, each wave owns one 16-query tile of one (b,h).
// K chunks: TDM tensor_load_to_lds (padded rows), double-buffered with
// s_wait_tensorcnt thresholds. V chunks: register ping-pong + transpose-stage.
// ---------------------------------------------------------------------------
__global__ __launch_bounds__(128)
void attn_softmax1_kernel(const _Float16* __restrict__ Qh,
                          const _Float16* __restrict__ Kh,
                          const _Float16* __restrict__ Vh,
                          const int* __restrict__ mask,
                          _Float16* __restrict__ Xout)
{
    constexpr int KL = 72;   // K tile row stride: 64 dk + 8 pad (TDM pad: 4 DW / 32 DW)
    constexpr int VL = 40;   // V^T row stride   (32 kv + pad)
    constexpr int PL = 40;   // P row stride     (32 kv + pad)
    __shared__ alignas(16) _Float16 Ks [2][32 * KL];   // K chunk [kv][dk]
    __shared__ alignas(16) _Float16 Vts[2][64 * VL];   // V chunk transposed [dk][kv]
    __shared__ alignas(16) _Float16 Ps [4][16 * PL];   // per-wave P staging [q][kv]

    const int tid  = threadIdx.x;
    const int lane = tid & 31;
    const int wave = tid >> 5;
    const int hh   = blockIdx.y;
    const int bb   = blockIdx.z;
    const int q0   = (blockIdx.x * 4 + wave) * 16;

    const size_t headOff = ((size_t)(bb * Hc + hh)) * Sc * DKc;
    const _Float16* Qp = Qh + headOff;
    const _Float16* Kp = Kh + headOff;
    const _Float16* Vp = Vh + headOff;
    const int*      mp = mask + bb * Sc;

    const int mloc = lane & 15;
    const int half = lane >> 4;

    // Q fragments (A layout), dk chunks 0-31 and 32-63; loop-invariant.
    v16h qa[2];
    #pragma unroll
    for (int g = 0; g < 2; ++g) {
        const _Float16* qrow = Qp + (size_t)(q0 + mloc) * DKc + g * 32 + half * 8;
        *((v8h*)&qa[g] + 0) = *(const v8h*)(qrow + 0);
        *((v8h*)&qa[g] + 1) = *(const v8h*)(qrow + 16);
    }

    v8f o[4] = {{}, {}, {}, {}};             // O accum: 4 dk-tiles of 16x16
    float mrow[8], lrow[8];
    #pragma unroll
    for (int r = 0; r < 8; ++r) { mrow[r] = -1e30f; lrow[r] = 0.0f; }

    // staging: 128 threads, 32 rows x 64 cols, 16 f16 per thread
    const int krow = tid >> 2;
    const int kcol = (tid & 3) * 16;

    // ---- preload chunk 0: TDM for K, registers for V ----
    if (wave == 0) {
        tdm_load_2d_f16((unsigned)(unsigned long long)&Ks[0][0], Kp,
                        /*tensor_d0=*/DKc, /*tensor_d1=*/Sc,
                        /*tile_d0=*/DKc, /*tile_d1=*/32, /*d0_stride=*/DKc,
                        /*pad_interval(32DW)=*/4, /*pad_amount(4DW)=*/3, true);
    }
    v8h v0r, v1r;
    {
        const _Float16* src = Vp + (size_t)krow * DKc + kcol;
        v0r = ((const v8h*)src)[0];
        v1r = ((const v8h*)src)[1];
    }

    const int NCH = Sc / 32;   // 64 chunks
    for (int c = 0; c < NCH; ++c) {
        const int cur = c & 1;
        const int nxt = cur ^ 1;
        const int kv0 = c * 32;

        // commit V chunk c (registers) transposed into Vts[cur]
        #pragma unroll
        for (int e = 0; e < 8; ++e) Vts[cur][(kcol + e) * VL + krow]     = v0r[e];
        #pragma unroll
        for (int e = 0; e < 8; ++e) Vts[cur][(kcol + 8 + e) * VL + krow] = v1r[e];

        // kick off chunk c+1
        if (c + 1 < NCH) {
            if (wave == 0) {
                tdm_load_2d_f16((unsigned)(unsigned long long)&Ks[nxt][0],
                                Kp + (size_t)(kv0 + 32) * DKc,
                                DKc, Sc, DKc, 32, DKc, 4, 3, true);
            }
            const _Float16* src = Vp + (size_t)(kv0 + 32 + krow) * DKc + kcol;
            v0r = ((const v8h*)src)[0];
            v1r = ((const v8h*)src)[1];
        }

        // wait for chunk c's TDM; chunk c+1's stays in flight (in-order per wave)
        if (c + 1 < NCH) __builtin_amdgcn_s_wait_tensorcnt(1);
        else             __builtin_amdgcn_s_wait_tensorcnt(0);
        __syncthreads();

        // ---- scores: two 16(q) x 16(kv) tiles, contract over dk=64 ----
        v8f s0 = {}, s1 = {};
        #pragma unroll
        for (int g = 0; g < 2; ++g) {
            v16h b0, b1;
            const _Float16* k0r = &Ks[cur][(0  + mloc) * KL + g * 32 + half * 8];
            *((v8h*)&b0 + 0) = *(const v8h*)(k0r + 0);
            *((v8h*)&b0 + 1) = *(const v8h*)(k0r + 16);
            const _Float16* k1r = &Ks[cur][(16 + mloc) * KL + g * 32 + half * 8];
            *((v8h*)&b1 + 0) = *(const v8h*)(k1r + 0);
            *((v8h*)&b1 + 1) = *(const v8h*)(k1r + 16);
            s0 = WMMA_F32_F16(qa[g], b0, s0);
            s1 = WMMA_F32_F16(qa[g], b1, s1);
        }

        // ---- scale + mask (mask indexed by kv position = N = lane%16) ----
        const float scale = 0.125f;   // 1/sqrt(64)
        const int mk0 = mp[kv0 + mloc];
        const int mk1 = mp[kv0 + 16 + mloc];
        #pragma unroll
        for (int r = 0; r < 8; ++r) {
            float a0 = s0[r] * scale; if (mk0 == 0) a0 = -1.0e9f;
            float a1 = s1[r] * scale; if (mk1 == 0) a1 = -1.0e9f;
            s0[r] = a0; s1[r] = a1;
        }

        // ---- online softmax-1 per row (row M = half*8 + r, across 16 lanes) ----
        float alpha[8];
        #pragma unroll
        for (int r = 0; r < 8; ++r) {
            float mx = fmaxf(s0[r], s1[r]);
            #pragma unroll
            for (int off = 1; off < 16; off <<= 1)
                mx = fmaxf(mx, __shfl_xor(mx, off, 32));
            float mn = fmaxf(mrow[r], mx);
            float al = __expf(mrow[r] - mn);
            float p0 = __expf(s0[r] - mn);
            float p1 = __expf(s1[r] - mn);
            float sum = p0 + p1;
            #pragma unroll
            for (int off = 1; off < 16; off <<= 1)
                sum += __shfl_xor(sum, off, 32);
            lrow[r]  = lrow[r] * al + sum;
            mrow[r]  = mn;
            alpha[r] = al;
            s0[r] = p0; s1[r] = p1;
        }
        #pragma unroll
        for (int t = 0; t < 4; ++t)
            #pragma unroll
            for (int r = 0; r < 8; ++r) o[t][r] *= alpha[r];

        // ---- P: C-layout -> LDS row-major [q][kv] (wave-private region) ----
        _Float16* Pw = Ps[wave];
        #pragma unroll
        for (int r = 0; r < 8; ++r) {
            int mr = half * 8 + r;
            Pw[mr * PL + mloc]      = (_Float16)s0[r];
            Pw[mr * PL + 16 + mloc] = (_Float16)s1[r];
        }
        __syncthreads();   // also orders same-wave DS write->read

        // ---- P*V: A = P (16x32 over kv), B = V^T tiles (32kv x 16dk) ----
        v16h pa;
        const _Float16* prow = Pw + mloc * PL + half * 8;
        *((v8h*)&pa + 0) = *(const v8h*)(prow + 0);
        *((v8h*)&pa + 1) = *(const v8h*)(prow + 16);
        #pragma unroll
        for (int t = 0; t < 4; ++t) {
            v16h vb;
            const _Float16* vrow = &Vts[cur][(t * 16 + mloc) * VL + half * 8];
            *((v8h*)&vb + 0) = *(const v8h*)(vrow + 0);
            *((v8h*)&vb + 1) = *(const v8h*)(vrow + 16);
            o[t] = WMMA_F32_F16(pa, vb, o[t]);
        }
        __syncthreads();   // buffer `cur` fully consumed before refill at c+2
    }

    // ---- finalize: out = O / (1 + l), write f16 (B,S,D) layout ----
    #pragma unroll
    for (int r = 0; r < 8; ++r) {
        float inv = 1.0f / (1.0f + lrow[r]);
        int q = q0 + half * 8 + r;
        #pragma unroll
        for (int t = 0; t < 4; ++t) {
            int dk = t * 16 + mloc;
            Xout[((size_t)(bb * Sc + q)) * Dc + hh * DKc + dk] =
                (_Float16)(o[t][r] * inv);
        }
    }
}

// ---------------------------------------------------------------------------
extern "C" void kernel_launch(void* const* d_in, const int* in_sizes, int n_in,
                              void* d_out, int out_size, void* d_ws, size_t ws_size,
                              hipStream_t stream)
{
    const float* query = (const float*)d_in[0];
    const float* key   = (const float*)d_in[1];
    const float* value = (const float*)d_in[2];
    const int*   mask  = (const int*)  d_in[3];
    const float* Wq    = (const float*)d_in[4];
    const float* bq    = (const float*)d_in[5];
    const float* Wk    = (const float*)d_in[6];
    const float* bk    = (const float*)d_in[7];
    const float* Wv    = (const float*)d_in[8];
    const float* bv    = (const float*)d_in[9];
    const float* Wo    = (const float*)d_in[10];
    const float* bo    = (const float*)d_in[11];

    // workspace: Qh/Kh/Vh/Xh f16 (8 MB each) + 4 f16 weight matrices (2 MB each)
    char* ws = (char*)d_ws;
    const size_t bsd = (size_t)Bc * Sc * Dc * sizeof(_Float16);   // 8 MB
    const size_t wsz = (size_t)Dc * Dc * sizeof(_Float16);        // 2 MB
    _Float16* Qh  = (_Float16*)(ws);
    _Float16* Kh  = (_Float16*)(ws + 1 * bsd);
    _Float16* Vh  = (_Float16*)(ws + 2 * bsd);
    _Float16* Xh  = (_Float16*)(ws + 3 * bsd);
    _Float16* Wq6 = (_Float16*)(ws + 4 * bsd);
    _Float16* Wk6 = (_Float16*)(ws + 4 * bsd + 1 * wsz);
    _Float16* Wv6 = (_Float16*)(ws + 4 * bsd + 2 * wsz);
    _Float16* Wo6 = (_Float16*)(ws + 4 * bsd + 3 * wsz);

    // 0) convert weights to f16 once
    const int nW = Dc * Dc;
    dim3 cgrid(nW / (256 * 8));
    cvt_f32_to_f16_kernel<<<cgrid, 256, 0, stream>>>(Wq, Wq6, nW);
    cvt_f32_to_f16_kernel<<<cgrid, 256, 0, stream>>>(Wk, Wk6, nW);
    cvt_f32_to_f16_kernel<<<cgrid, 256, 0, stream>>>(Wv, Wv6, nW);
    cvt_f32_to_f16_kernel<<<cgrid, 256, 0, stream>>>(Wo, Wo6, nW);

    const int M = Bc * Sc;                 // 4096
    dim3 pgrid(Dc / 64, M / 32);           // (16, 128)
    dim3 pblock(256);

    // 1) QKV projections (A = f32 input, head-layout f16 output)
    proj_gemm_kernel<false, true><<<pgrid, pblock, 0, stream>>>(
        query, nullptr, Wq6, bq, Qh, nullptr, M, Dc, Dc);
    proj_gemm_kernel<false, true><<<pgrid, pblock, 0, stream>>>(
        key,   nullptr, Wk6, bk, Kh, nullptr, M, Dc, Dc);
    proj_gemm_kernel<false, true><<<pgrid, pblock, 0, stream>>>(
        value, nullptr, Wv6, bv, Vh, nullptr, M, Dc, Dc);

    // 2) attention
    dim3 agrid(Sc / 64, Hc, Bc);           // (32, 16, 2); 4 q-tiles per block
    attn_softmax1_kernel<<<agrid, dim3(128), 0, stream>>>(Qh, Kh, Vh, mask, Xh);

    // 3) output projection (A = X f16 async, f32 flat output)
    proj_gemm_kernel<true, false><<<pgrid, pblock, 0, stream>>>(
        nullptr, Xh, Wo6, bo, nullptr, (float*)d_out, M, Dc, Dc);
}
